// LightningGNN_52055003628016
// MI455X (gfx1250) — matrix operand order repaired and checked by reading, not compiled, not run on hardware
//
#include <hip/hip_runtime.h>

typedef __attribute__((ext_vector_type(16))) _Float16 v16h;
typedef __attribute__((ext_vector_type(8)))  _Float16 v8h;
typedef __attribute__((ext_vector_type(8)))  float    v8f;
typedef __attribute__((ext_vector_type(4)))  unsigned int v4u;
typedef __attribute__((ext_vector_type(4)))  int      v4i;
typedef __attribute__((ext_vector_type(8)))  int      v8i;

#define T_IN    512
#define H_DIM   64
#define G_GRP   512
#define ENABLE_TDM 1

__device__ __forceinline__ v16h cat8(v8h lo, v8h hi)
{
    return __builtin_shufflevector(lo, hi, 0, 1, 2, 3, 4, 5, 6, 7,
                                   8, 9, 10, 11, 12, 13, 14, 15);
}

// ---------------------------------------------------------------------------
// Encoder: conv1 (1->16,k7,s2,p3) scalar -> conv2 (16->32,k5,s2,p2) WMMA
//          -> conv3 (32->64,k5,s2,p2) WMMA -> mean over time.
// 64 threads = 2 wave32; 2 nodes per block, one wave per node for WMMA.
// im2col activations stored [t][K] (K-contiguous): every WMMA B-fragment is
// one aligned 32B LDS read. B fragments are software-pipelined one tile
// ahead so s_wait_dscnt is hidden under the previous tile's WMMA chain.
// ---------------------------------------------------------------------------
__global__ __launch_bounds__(64) void encoder_kernel(
    const float* __restrict__ x,
    const float* __restrict__ w1, const float* __restrict__ b1,
    const float* __restrict__ w2, const float* __restrict__ b2,
    const float* __restrict__ w3, const float* __restrict__ b3,
    float* __restrict__ h_out, int n_nodes)
{
    __shared__ __align__(32) float    sx[2][512];
    __shared__ __align__(32) _Float16 sB1[2][128][96];   // conv2 im2col: [t][K=ic+16*kk] (pad K to 96)
    __shared__ __align__(32) _Float16 sB2[2][64][160];   // conv3 im2col: [t][K=ic+32*kk] (K=160 exact)
    __shared__ __align__(32) _Float16 sw2[32][96];       // conv2 A: [oc][K]
    __shared__ __align__(32) _Float16 sw3[64][160];      // conv3 A: [oc][K]
    __shared__ float sw1[16 * 7];
    __shared__ float sb1[16], sb2[32], sb3[64];

    const int tid = threadIdx.x;
    const int nb  = blockIdx.x * 2;

    // ---- stage weights (im2col A matrices) ----
    for (int i = tid; i < 16 * 7; i += 64) sw1[i] = w1[i];
    if (tid < 16) sb1[tid] = b1[tid];
    if (tid < 32) sb2[tid] = b2[tid];
    if (tid < 64) sb3[tid] = b3[tid];
    for (int i = tid; i < 32 * 96; i += 64) {
        int oc = i / 96, k = i % 96, ic = k & 15, kk = k >> 4;
        sw2[oc][k] = (_Float16)((kk < 5) ? w2[(oc * 16 + ic) * 5 + kk] : 0.f);
    }
    for (int i = tid; i < 64 * 160; i += 64) {
        int oc = i / 160, k = i % 160, ic = k & 31, kk = k >> 5;
        sw3[oc][k] = (_Float16)w3[(oc * 32 + ic) * 5 + kk];
    }

    // ---- zero only the K-padding + conv halo slots ----
    {
        v16h z = {};
        for (int i = tid; i < 2 * 128; i += 64) {     // sB1 K-pad cols 80..95
            int j = i >> 7, t = i & 127;
            *(v16h*)&sB1[j][t][80] = z;
        }
        if (tid < 2) {
            int j = tid;
            *(v16h*)&sB1[j][0][0]    = z;  // t=0, kk=0  (p=-2)
            *(v16h*)&sB1[j][0][16]   = z;  // t=0, kk=1  (p=-1)
            *(v16h*)&sB1[j][127][64] = z;  // t=127, kk=4 (p=256)
            *(v16h*)&sB2[j][0][0]    = z;  // t=0, kk=0 (p=-2)
            *(v16h*)&sB2[j][0][16]   = z;
            *(v16h*)&sB2[j][0][32]   = z;  // t=0, kk=1 (p=-1)
            *(v16h*)&sB2[j][0][48]   = z;
            *(v16h*)&sB2[j][63][128] = z;  // t=63, kk=4 (p=128)
            *(v16h*)&sB2[j][63][144] = z;
        }
    }

    // ---- stage x: TDM (Tensor Data Mover) if available, else plain loads ----
#if ENABLE_TDM && defined(__has_builtin) && \
    __has_builtin(__builtin_amdgcn_tensor_load_to_lds) && \
    __has_builtin(__builtin_amdgcn_s_wait_tensorcnt)
    if (tid < 32) {                       // one wave issues the DMA descriptor
        unsigned long long ga = (unsigned long long)(uintptr_t)(x + (size_t)nb * T_IN);
        unsigned lds = (unsigned)(uintptr_t)(void*)&sx[0][0];
        v4u g0;                           // D# group 0 (ISA 8.3)
        g0.x = 1u;                                        // count=1, user desc
        g0.y = lds;                                       // lds_addr
        g0.z = (unsigned)ga;                              // global_addr[31:0]
        g0.w = ((unsigned)(ga >> 32) & 0x1FFFFFFu) | (2u << 30); // addr[56:32] | type=2
        v8i g1;                           // D# group 1 (ISA 8.4): 2D tensor, 4B elems
        g1[0] = (int)(2u << 16);                          // wg_mask=0, data_size=2 (4B)
        g1[1] = (int)(512u << 16);                        // tensor_dim0 = 512
        g1[2] = (int)(((unsigned)n_nodes & 0xFFFFu) << 16); // tensor_dim1 lo
        g1[3] = (int)(((unsigned)n_nodes >> 16) | (512u << 16)); // dim1 hi | tile_dim0=512
        g1[4] = 2;                                        // tile_dim1=2 rows
        g1[5] = 512;                                      // tensor_dim0_stride
        g1[6] = 0;
        g1[7] = 0;
        v4i zg; zg[0] = 0; zg[1] = 0; zg[2] = 0; zg[3] = 0;
#if __clang_major__ >= 23
        v8i zg8; zg8[0]=0; zg8[1]=0; zg8[2]=0; zg8[3]=0; zg8[4]=0; zg8[5]=0; zg8[6]=0; zg8[7]=0;
        __builtin_amdgcn_tensor_load_to_lds(g0, g1, zg, zg, zg8, 0);
#else
        __builtin_amdgcn_tensor_load_to_lds(g0, g1, zg, zg, 0);
#endif
        __builtin_amdgcn_s_wait_tensorcnt(0);
    }
#else
    for (int i = tid; i < 2 * 512; i += 64) {
        int j = i >> 9, t = i & 511, n = nb + j;
        sx[j][t] = (n < n_nodes) ? x[(size_t)n * T_IN + t] : 0.f;
    }
#endif
    __syncthreads();

    // ---- conv1: each thread computes all 16 channels of an output pair
    //      (p0=2q, p1=2q+1) and scatters whole v16h rows into conv2 im2col ----
    for (int i = tid; i < 2 * 128; i += 64) {
        int j = i >> 7, q = i & 127;
        float xv[9];                         // input positions 4q-3 .. 4q+5
#pragma unroll
        for (int u = 0; u < 9; ++u) {
            int pos = 4 * q - 3 + u;
            xv[u] = (pos >= 0 && pos < 512) ? sx[j][pos] : 0.f;
        }
        v16h row0, row1;
#pragma unroll
        for (int ic = 0; ic < 16; ++ic) {
            float a0 = sb1[ic], a1 = sb1[ic];
#pragma unroll
            for (int kk = 0; kk < 7; ++kk) {
                float w = sw1[ic * 7 + kk];
                a0 += w * xv[kk];            // p0: input = 4q-3+kk
                a1 += w * xv[kk + 2];        // p1: input = 4q-1+kk
            }
            row0[ic] = (_Float16)fmaxf(a0, 0.f);
            row1[ic] = (_Float16)fmaxf(a1, 0.f);
        }
        if (q < 127) {
            *(v16h*)&sB1[j][q + 1][0]  = row0;   // kk=0
            *(v16h*)&sB1[j][q + 1][16] = row1;   // kk=1
        }
        *(v16h*)&sB1[j][q][32] = row0;           // kk=2
        *(v16h*)&sB1[j][q][48] = row1;           // kk=3
        if (q >= 1)
            *(v16h*)&sB1[j][q - 1][64] = row0;   // kk=4
    }
    __syncthreads();

    const int wid = tid >> 5, lane = tid & 31;
    const int hf = lane >> 4, ln = lane & 15;
    const int j = wid;                      // this wave's node slot

    // ---- conv2 via WMMA: out 32x128, K=80 padded to 96 (3 steps);
    //      B fragments double-buffered across nt tiles ----
    for (int mt = 0; mt < 2; ++mt) {
        v16h afr[3];
        float bias2[8];
#pragma unroll
        for (int ks = 0; ks < 3; ++ks) {
            v8h alo = *(const v8h*)&sw2[16 * mt + ln][32 * ks + 16 * hf];
            v8h ahi = *(const v8h*)&sw2[16 * mt + ln][32 * ks + 16 * hf + 8];
            afr[ks] = cat8(alo, ahi);
        }
#pragma unroll
        for (int r = 0; r < 8; ++r) bias2[r] = sb2[16 * mt + 8 * hf + r];

        v16h bcur[3];
#pragma unroll
        for (int ks = 0; ks < 3; ++ks)
            bcur[ks] = *(const v16h*)&sB1[j][ln][32 * ks + 16 * hf];
        for (int nt = 0; nt < 8; ++nt) {
            v16h bnxt[3] = {};
            if (nt < 7) {
#pragma unroll
                for (int ks = 0; ks < 3; ++ks)
                    bnxt[ks] = *(const v16h*)&sB1[j][16 * (nt + 1) + ln][32 * ks + 16 * hf];
            }
            v8f acc = {};
#pragma unroll
            for (int ks = 0; ks < 3; ++ks)
                acc = __builtin_amdgcn_wmma_f32_16x16x32_f16(
                    false, afr[ks], false, bcur[ks], (short)0, acc, false, false);
            // lane holds 8 consecutive output channels for position p
            v8h vv;
#pragma unroll
            for (int r = 0; r < 8; ++r)
                vv[r] = (_Float16)fmaxf(acc[r] + bias2[r], 0.f);
            int p = 16 * nt + ln, q = p >> 1;
            int oc0 = 16 * mt + 8 * hf;
            if ((p & 1) == 0) {              // kk = 0,2,4
                if (q < 63) *(v8h*)&sB2[j][q + 1][oc0]       = vv;
                *(v8h*)&sB2[j][q][64 + oc0]                  = vv;
                if (q >= 1) *(v8h*)&sB2[j][q - 1][128 + oc0] = vv;
            } else {                         // kk = 1,3
                if (q < 63) *(v8h*)&sB2[j][q + 1][32 + oc0]  = vv;
                *(v8h*)&sB2[j][q][96 + oc0]                  = vv;
            }
#pragma unroll
            for (int ks = 0; ks < 3; ++ks) bcur[ks] = bnxt[ks];
        }
    }

    // ---- conv3 via WMMA: out 64x64, K=160 (5 steps); relu then mean over t;
    //      B fragments double-buffered across nt tiles ----
    for (int mt = 0; mt < 4; ++mt) {
        v16h afr[5];
        float bias3[8];
#pragma unroll
        for (int ks = 0; ks < 5; ++ks) {
            v8h alo = *(const v8h*)&sw3[16 * mt + ln][32 * ks + 16 * hf];
            v8h ahi = *(const v8h*)&sw3[16 * mt + ln][32 * ks + 16 * hf + 8];
            afr[ks] = cat8(alo, ahi);
        }
#pragma unroll
        for (int r = 0; r < 8; ++r) bias3[r] = sb3[16 * mt + 8 * hf + r];

        float rs[8];
#pragma unroll
        for (int r = 0; r < 8; ++r) rs[r] = 0.f;

        v16h bcur[5];
#pragma unroll
        for (int ks = 0; ks < 5; ++ks)
            bcur[ks] = *(const v16h*)&sB2[j][ln][32 * ks + 16 * hf];
        for (int nt = 0; nt < 4; ++nt) {
            v16h bnxt[5] = {};
            if (nt < 3) {
#pragma unroll
                for (int ks = 0; ks < 5; ++ks)
                    bnxt[ks] = *(const v16h*)&sB2[j][16 * (nt + 1) + ln][32 * ks + 16 * hf];
            }
            v8f acc = {};
#pragma unroll
            for (int ks = 0; ks < 5; ++ks)
                acc = __builtin_amdgcn_wmma_f32_16x16x32_f16(
                    false, afr[ks], false, bcur[ks], (short)0, acc, false, false);
#pragma unroll
            for (int r = 0; r < 8; ++r)
                rs[r] += fmaxf(acc[r] + bias3[r], 0.f);
#pragma unroll
            for (int ks = 0; ks < 5; ++ks) bcur[ks] = bnxt[ks];
        }
#pragma unroll
        for (int r = 0; r < 8; ++r) {        // sum the 16 lanes of each half
            float v = rs[r];
            v += __shfl_xor(v, 1);
            v += __shfl_xor(v, 2);
            v += __shfl_xor(v, 4);
            v += __shfl_xor(v, 8);
            rs[r] = v;
        }
        int n = nb + j;
        if (ln == 0 && n < n_nodes) {
#pragma unroll
            for (int r = 0; r < 8; ++r)
                h_out[(size_t)n * H_DIM + 16 * mt + 8 * hf + r] = rs[r] * (1.f / 64.f);
        }
    }
}

// ---------------------------------------------------------------------------
// hw = h @ W  (N x 64) @ (64 x 64), f16 WMMA / f32 accumulate.
// W staged transposed so B-fragments are K-contiguous 32B LDS reads;
// fragments double-buffered across the per-wave tile loop.
// ---------------------------------------------------------------------------
__global__ __launch_bounds__(256) void gemm_hW(
    const float* __restrict__ Hin, const float* __restrict__ W,
    float* __restrict__ Out, int nrows)
{
    __shared__ __align__(32) _Float16 sh[128][64];
    __shared__ __align__(32) _Float16 swT[64][64];   // swT[c][k] = W[k][c]
    const int tid  = threadIdx.x;
    const int base = blockIdx.x * 128;

    for (int i = tid; i < 64 * 64; i += 256) {
        int k = i >> 6, c = i & 63;
        swT[c][k] = (_Float16)W[i];
    }
    for (int i = tid; i < 128 * 64; i += 256) {
        int rl = i >> 6, c = i & 63, r = base + rl;
        sh[rl][c] = (_Float16)((r < nrows) ? Hin[(size_t)r * H_DIM + c] : 0.f);
    }
    __syncthreads();

    const int wid = tid >> 5, lane = tid & 31;
    const int hf = lane >> 4, ln = lane & 15;

    v16h afr[2], bfr[2];
    {
        int mt = wid >> 2, nt = wid & 3;     // first tile for this wave
#pragma unroll
        for (int ks = 0; ks < 2; ++ks) {
            v8h alo = *(const v8h*)&sh[16 * mt + ln][32 * ks + 16 * hf];
            v8h ahi = *(const v8h*)&sh[16 * mt + ln][32 * ks + 16 * hf + 8];
            afr[ks] = cat8(alo, ahi);
            bfr[ks] = *(const v16h*)&swT[16 * nt + ln][32 * ks + 16 * hf];
        }
    }
    for (int tile = wid; tile < 32; tile += 8) {
        v16h anxt[2] = {}, bnxt[2] = {};
        if (tile + 8 < 32) {
            int mt2 = (tile + 8) >> 2, nt2 = (tile + 8) & 3;
#pragma unroll
            for (int ks = 0; ks < 2; ++ks) {
                v8h alo = *(const v8h*)&sh[16 * mt2 + ln][32 * ks + 16 * hf];
                v8h ahi = *(const v8h*)&sh[16 * mt2 + ln][32 * ks + 16 * hf + 8];
                anxt[ks] = cat8(alo, ahi);
                bnxt[ks] = *(const v16h*)&swT[16 * nt2 + ln][32 * ks + 16 * hf];
            }
        }
        v8f acc = {};
#pragma unroll
        for (int ks = 0; ks < 2; ++ks)
            acc = __builtin_amdgcn_wmma_f32_16x16x32_f16(
                false, afr[ks], false, bfr[ks], (short)0, acc, false, false);
        int mt = tile >> 2, nt = tile & 3;
#pragma unroll
        for (int r = 0; r < 8; ++r) {
            int row = base + 16 * mt + r + 8 * hf;
            if (row < nrows) Out[(size_t)row * H_DIM + 16 * nt + ln] = acc[r];
        }
#pragma unroll
        for (int ks = 0; ks < 2; ++ks) { afr[ks] = anxt[ks]; bfr[ks] = bnxt[ks]; }
    }
}

// ---------------------------------------------------------------------------
// Graph / pooling helpers (hw/agg tables are 12.8 MB -> L2-resident atomics)
// ---------------------------------------------------------------------------
__global__ void zero_f32(float* __restrict__ p, int n)
{
    int i = blockIdx.x * blockDim.x + threadIdx.x;
    if (i < n) p[i] = 0.f;
}

__global__ void deg_kernel(const int* __restrict__ dst, float* __restrict__ deg, int ne)
{
    int i = blockIdx.x * blockDim.x + threadIdx.x;
    if (i < ne) atomicAdd(&deg[dst[i]], 1.f);
}

__global__ void dinv_kernel(float* __restrict__ dinv, const float* __restrict__ deg, int n)
{
    int i = blockIdx.x * blockDim.x + threadIdx.x;
    if (i < n) dinv[i] = rsqrtf(deg[i] + 1.f);
}

__global__ void norm_kernel(const int* __restrict__ src, const int* __restrict__ dst,
                            const float* __restrict__ dinv, float* __restrict__ nrm, int ne)
{
    int i = blockIdx.x * blockDim.x + threadIdx.x;
    if (i < ne) nrm[i] = dinv[src[i]] * dinv[dst[i]];
}

__global__ void agg_kernel(const float* __restrict__ hw,
                           const int* __restrict__ src, const int* __restrict__ dst,
                           const float* __restrict__ nrm, float* __restrict__ agg, int total)
{
    int i = blockIdx.x * blockDim.x + threadIdx.x;
    if (i >= total) return;
    int e = i >> 6, c = i & 63;
    int s = src[e], d = dst[e];
    atomicAdd(&agg[d * H_DIM + c], hw[s * H_DIM + c] * nrm[e]);
}

__global__ void finalize_kernel(const float* __restrict__ agg, const float* __restrict__ hw,
                                const float* __restrict__ dinv, const float* __restrict__ b,
                                float* __restrict__ out, int n64)
{
    int i = blockIdx.x * blockDim.x + threadIdx.x;
    if (i >= n64) return;
    int node = i >> 6, c = i & 63;
    float d = dinv[node];
    out[i] = fmaxf(agg[i] + hw[i] * d * d + b[c], 0.f);
}

__global__ void cnt_kernel(const int* __restrict__ batch, float* __restrict__ cnt, int n)
{
    int i = blockIdx.x * blockDim.x + threadIdx.x;
    if (i < n) atomicAdd(&cnt[batch[i]], 1.f);
}

__global__ void pool_kernel(const float* __restrict__ h, const int* __restrict__ batch,
                            float* __restrict__ pooled, int n64)
{
    int i = blockIdx.x * blockDim.x + threadIdx.x;
    if (i >= n64) return;
    int node = i >> 6, c = i & 63;
    atomicAdd(&pooled[batch[node] * H_DIM + c], h[i]);
}

__global__ void head_kernel(const float* __restrict__ pooled, const float* __restrict__ cnt,
                            const float* __restrict__ lW, const float* __restrict__ lb,
                            float* __restrict__ out)
{
    int i = blockIdx.x * blockDim.x + threadIdx.x;
    if (i >= G_GRP * 2) return;
    int g = i >> 1, o = i & 1;
    float inv = 1.f / fmaxf(cnt[g], 1.f);
    float s = 0.f;
#pragma unroll
    for (int c = 0; c < H_DIM; ++c) s += pooled[g * H_DIM + c] * lW[c * 2 + o];
    out[i] = s * inv + lb[o];
}

// ---------------------------------------------------------------------------
extern "C" void kernel_launch(void* const* d_in, const int* in_sizes, int n_in,
                              void* d_out, int out_size, void* d_ws, size_t ws_size,
                              hipStream_t stream)
{
    const float* x     = (const float*)d_in[0];
    const int*   ei    = (const int*)  d_in[1];
    const int*   batch = (const int*)  d_in[2];
    const float* w1    = (const float*)d_in[3];
    const float* b1    = (const float*)d_in[4];
    const float* w2    = (const float*)d_in[5];
    const float* b2    = (const float*)d_in[6];
    const float* w3    = (const float*)d_in[7];
    const float* b3    = (const float*)d_in[8];
    const float* gW1   = (const float*)d_in[9];
    const float* gb1   = (const float*)d_in[10];
    const float* gW2   = (const float*)d_in[11];
    const float* gb2   = (const float*)d_in[12];
    const float* lW    = (const float*)d_in[13];
    const float* lb    = (const float*)d_in[14];

    const int nn = in_sizes[0] / T_IN;     // 50000
    const int ne = in_sizes[1] / 2;        // 1600000
    const int*  srcI = ei;
    const int*  dstI = ei + ne;

    float* ws     = (float*)d_ws;
    float* h      = ws;                          // nn*64
    float* hw     = h   + (size_t)nn * H_DIM;    // nn*64
    float* agg    = hw  + (size_t)nn * H_DIM;    // nn*64
    float* dinv   = agg + (size_t)nn * H_DIM;    // nn
    float* deg    = dinv + nn;                   // nn
    float* nrm    = deg  + nn;                   // ne
    float* pooled = nrm  + ne;                   // G*64
    float* cnt    = pooled + (size_t)G_GRP * H_DIM; // G
    (void)n_in; (void)out_size; (void)ws_size;

    const int TPB = 256;
    const int n64 = nn * H_DIM;
    const int e64 = ne * H_DIM;

    // 1. encoder -> h
    encoder_kernel<<<(nn + 1) / 2, 64, 0, stream>>>(x, w1, b1, w2, b2, w3, b3, h, nn);

    // 2. degrees / norms (shared by both GCN layers)
    zero_f32<<<(nn + TPB - 1) / TPB, TPB, 0, stream>>>(deg, nn);
    deg_kernel<<<(ne + TPB - 1) / TPB, TPB, 0, stream>>>(dstI, deg, ne);
    dinv_kernel<<<(nn + TPB - 1) / TPB, TPB, 0, stream>>>(dinv, deg, nn);
    norm_kernel<<<(ne + TPB - 1) / TPB, TPB, 0, stream>>>(srcI, dstI, dinv, nrm, ne);

    // 3. GCN layer 1
    gemm_hW<<<(nn + 127) / 128, 256, 0, stream>>>(h, gW1, hw, nn);
    zero_f32<<<(n64 + TPB - 1) / TPB, TPB, 0, stream>>>(agg, n64);
    agg_kernel<<<(e64 + TPB - 1) / TPB, TPB, 0, stream>>>(hw, srcI, dstI, nrm, agg, e64);
    finalize_kernel<<<(n64 + TPB - 1) / TPB, TPB, 0, stream>>>(agg, hw, dinv, gb1, h, n64);

    // 4. GCN layer 2
    gemm_hW<<<(nn + 127) / 128, 256, 0, stream>>>(h, gW2, hw, nn);
    zero_f32<<<(n64 + TPB - 1) / TPB, TPB, 0, stream>>>(agg, n64);
    agg_kernel<<<(e64 + TPB - 1) / TPB, TPB, 0, stream>>>(hw, srcI, dstI, nrm, agg, e64);
    finalize_kernel<<<(n64 + TPB - 1) / TPB, TPB, 0, stream>>>(agg, hw, dinv, gb2, h, n64);

    // 5. mean pooling + head
    zero_f32<<<(G_GRP * H_DIM + TPB - 1) / TPB, TPB, 0, stream>>>(pooled, G_GRP * H_DIM);
    zero_f32<<<(G_GRP + TPB - 1) / TPB, TPB, 0, stream>>>(cnt, G_GRP);
    cnt_kernel<<<(nn + TPB - 1) / TPB, TPB, 0, stream>>>(batch, cnt, nn);
    pool_kernel<<<(n64 + TPB - 1) / TPB, TPB, 0, stream>>>(h, batch, pooled, n64);
    head_kernel<<<(G_GRP * 2 + TPB - 1) / TPB, TPB, 0, stream>>>(pooled, cnt, lW, lb, (float*)d_out);
}
